// NormalizingFlow_26087631356690
// MI455X (gfx1250) — matrix operand order repaired and checked by reading, not compiled, not run on hardware
//
#include <hip/hip_runtime.h>

typedef __attribute__((ext_vector_type(16))) _Float16 v16h;
typedef __attribute__((ext_vector_type(8)))  float    v8f;

#define NFLOW 8
#define HDIM  128
#define NSAMP 131072

// gfx1250 hardware tanh (TRANS op), computed in pairs: the second v_tanh is
// the required "1 independent op" spacing for the first; one trailing v_nop
// covers the second before its result is consumed.
__device__ __forceinline__ void tanh2_hw(float xa, float xb, float& ra, float& rb) {
  asm("v_tanh_f32 %0, %2\n\t"
      "v_tanh_f32 %1, %3\n\t"
      "v_nop"
      : "=&v"(ra), "=v"(rb)
      : "v"(xa), "v"(xb));
}

__launch_bounds__(256, 4)
__global__ void nf_flow_kernel(const float* __restrict__ eps,
                               const float* __restrict__ L_tril,
                               const float* __restrict__ base_mean,
                               const float* __restrict__ sW1, const float* __restrict__ sb1,
                               const float* __restrict__ sW2e, const float* __restrict__ sb2e,
                               const float* __restrict__ sW2o, const float* __restrict__ sb2o,
                               const float* __restrict__ tW1, const float* __restrict__ tb1,
                               const float* __restrict__ tW2e, const float* __restrict__ tb2e,
                               const float* __restrict__ tW2o, const float* __restrict__ tb2o,
                               float* __restrict__ out)
{
  // W1/b1 pre-permuted into per-lane WMMA-A consumption order:
  // [flow][net][chunk][hi][16 values in hh order]  (2048 floats each, 8 KB)
  __shared__ __align__(16) float Wpk[NFLOW*2*4*2*16];
  __shared__ __align__(16) float Bpk[NFLOW*2*4*2*16];
  __shared__ __align__(32) _Float16 W2h[NFLOW*2*2*HDIM]; // [flow][net][row][k] f16
  __shared__ __align__(32) _Float16 Hzero[HDIM];
  __shared__ __align__(16) float B2[NFLOW][4];           // s0,s1,t0,t1 biases
  __shared__ __align__(16) float stage[8][16][4];        // per wave C extraction

  const int tid = threadIdx.x;

  // ---- stage W1/b1 permuted into LDS ----
  for (int i = tid; i < NFLOW*2*4*2*16; i += blockDim.x) {
    const int hh    = i & 15;
    const int hi_   = (i >> 4) & 1;
    const int chunk = (i >> 5) & 3;
    const int net   = (i >> 7) & 1;
    const int flow  = i >> 8;
    const int k     = chunk*32 + hi_*16 + hh;
    const int src   = flow*HDIM + k;
    Wpk[i] = net ? tW1[src] : sW1[src];
    Bpk[i] = net ? tb1[src] : sb1[src];
  }
  // ---- W2 rows as f16 B-operand columns ----
  for (int i = tid; i < NFLOW*2*2*HDIM; i += blockDim.x) {
    const int k = i & (HDIM-1); const int row = (i >> 7) & 1;
    const int net = (i >> 8) & 1; const int flow = i >> 9;
    float v;
    if ((flow & 1) == 0) {
      const int e = flow >> 1;
      const float* src = net ? tW2e : sW2e;
      v = src[(e*2 + row)*HDIM + k];
    } else {
      const int o = flow >> 1;
      if (row == 0) { const float* src = net ? tW2o : sW2o; v = src[o*HDIM + k]; }
      else v = 0.0f;
    }
    W2h[i] = (_Float16)v;
  }
  for (int i = tid; i < HDIM; i += blockDim.x) Hzero[i] = (_Float16)0.0f;
  for (int i = tid; i < NFLOW*4; i += blockDim.x) {
    const int flow = i >> 2, c = i & 3; float v = 0.0f;
    if ((flow & 1) == 0) {
      const int e = flow >> 1;
      v = (c==0) ? sb2e[e*2] : (c==1) ? sb2e[e*2+1] : (c==2) ? tb2e[e*2] : tb2e[e*2+1];
    } else {
      const int o = flow >> 1;
      v = (c==0) ? sb2o[o] : (c==2) ? tb2o[o] : 0.0f;
    }
    B2[flow][c] = v;
  }
  __syncthreads();

  // ---- 3x3 cholesky of (tril(L)+1e-6 I)(tril(L)+1e-6 I)^T, per-thread ----
  float l00 = L_tril[0] + 1e-6f, l10 = L_tril[3], l11 = L_tril[4] + 1e-6f;
  float l20 = L_tril[6], l21 = L_tril[7], l22 = L_tril[8] + 1e-6f;
  float c00 = l00*l00;
  float c10 = l10*l00;
  float c11 = l10*l10 + l11*l11;
  float c20 = l20*l00;
  float c21 = l20*l10 + l21*l11;
  float c22 = l20*l20 + l21*l21 + l22*l22;
  float q00 = sqrtf(c00);
  float q10 = c10 / q00, q20 = c20 / q00;
  float q11 = sqrtf(c11 - q10*q10);
  float q21 = (c21 - q20*q10) / q11;
  float q22 = sqrtf(c22 - q20*q20 - q21*q21);
  const float bm0 = base_mean[0], bm1 = base_mean[1], bm2 = base_mean[2];

  const int lane = tid & 31;
  const int m    = lane & 15;     // sample-in-tile (shared by lane pair)
  const int hi   = lane >> 4;     // K-split half held by this lane
  const int wave = tid >> 5;
  const int waveGlobal = blockIdx.x * (blockDim.x >> 5) + wave;
  const int numWaves   = gridDim.x * (blockDim.x >> 5);
  const int nTiles     = NSAMP / 16;

  const float4* W4 = (const float4*)Wpk;
  const float4* Q4 = (const float4*)Bpk;

  for (int tile = waveGlobal; tile < nTiles; tile += numWaves) {
    const int s = tile*16 + m;
    // speculative prefetch of the next tile's eps row for this lane
    __builtin_prefetch(eps + 3*(s + numWaves*16), 0, 0);

    float e0 = eps[s*3+0], e1 = eps[s*3+1], e2 = eps[s*3+2];
    float z0 = bm0 + q00*e0;
    float z1 = bm1 + q10*e0 + q11*e1;
    float z2 = bm2 + q20*e0 + q21*e1 + q22*e2;
    float logdet = 0.0f;

    for (int flow = 0; flow < NFLOW; ++flow) {
      const bool even = ((flow & 1) == 0);
      const float x1  = even ? z0 : z2;
      const int nrows = even ? 2 : 1;

      #pragma unroll
      for (int net = 0; net < 2; ++net) {
        // B operand: lane m supplies column m = W2 row m (zero beyond valid rows)
        const _Float16* brow = (m < nrows) ? &W2h[((flow*2 + net)*2 + m)*HDIM]
                                           : &Hzero[0];
        v8f acc = {};
        #pragma unroll
        for (int chunk = 0; chunk < 4; ++chunk) {
          // 16 contiguous pre-permuted W/b values for this lane: 8x ds_load_b128
          const int fb = (((flow*2 + net)*4 + chunk)*2 + hi) * 4; // float4 index
          float wv[16], bv[16];
          #pragma unroll
          for (int q = 0; q < 4; ++q) {
            ((float4*)wv)[q] = W4[fb + q];
            ((float4*)bv)[q] = Q4[fb + q];
          }
          v16h a;
          #pragma unroll
          for (int p = 0; p < 8; ++p) {
            const float ya = fmaf(x1, wv[2*p],   bv[2*p]);
            const float yb = fmaf(x1, wv[2*p+1], bv[2*p+1]);
            float ra, rb;
            tanh2_hw(ya, yb, ra, rb);
            a[2*p]   = (_Float16)ra;
            a[2*p+1] = (_Float16)rb;
          }
          const v16h b = *(const v16h*)(brow + chunk*32 + hi*16);
          acc = __builtin_amdgcn_wmma_f32_16x16x32_f16(
                  false, a, false, b, (short)0, acc, false, false);
        }
        if (m < 2) {            // columns 0..1 of C are the net outputs
          #pragma unroll
          for (int j = 0; j < 8; ++j)
            stage[wave][j + hi*8][net*2 + m] = acc[j];   // row M = j + hi*8
        }
      }

      // per-wave DS ops are in program order: reads below see the stores above
      const float4 st = *(const float4*)&stage[wave][m][0];
      const float4 bb = *(const float4*)&B2[flow][0];
      const float s0 = st.x + bb.x;
      const float s1 = st.y + bb.y;
      const float t0 = st.z + bb.z;
      const float t1 = st.w + bb.w;

      if (even) {               // x1=z0, x2=(z1,z2), s/t have 2 outputs
        z1 = z1*__expf(s0) + t0;
        z2 = z2*__expf(s1) + t1;
        logdet += s0 + s1;
      } else {                  // x1=z2, x2=(z0,z1), s/t broadcast 1 output
        const float es = __expf(s0);
        z0 = z0*es + t0;
        z1 = z1*es + t0;
        logdet += s0;
      }
    }

    if (hi == 0) {
      out[s*3+0] = z0; out[s*3+1] = z1; out[s*3+2] = z2;
      out[NSAMP*3 + s] = logdet;
    }
  }
}

extern "C" void kernel_launch(void* const* d_in, const int* in_sizes, int n_in,
                              void* d_out, int out_size, void* d_ws, size_t ws_size,
                              hipStream_t stream) {
  (void)in_sizes; (void)n_in; (void)out_size; (void)d_ws; (void)ws_size;
  const float* eps       = (const float*)d_in[0];
  const float* L_tril    = (const float*)d_in[1];
  const float* base_mean = (const float*)d_in[2];
  const float* sW1  = (const float*)d_in[3];
  const float* sb1  = (const float*)d_in[4];
  const float* sW2e = (const float*)d_in[5];
  const float* sb2e = (const float*)d_in[6];
  const float* sW2o = (const float*)d_in[7];
  const float* sb2o = (const float*)d_in[8];
  const float* tW1  = (const float*)d_in[9];
  const float* tb1  = (const float*)d_in[10];
  const float* tW2e = (const float*)d_in[11];
  const float* tb2e = (const float*)d_in[12];
  const float* tW2o = (const float*)d_in[13];
  const float* tb2o = (const float*)d_in[14];
  float* out = (float*)d_out;

  dim3 grid(1024), block(256);   // 8192 waves -> one 16-sample tile each
  nf_flow_kernel<<<grid, block, 0, stream>>>(eps, L_tril, base_mean,
                                             sW1, sb1, sW2e, sb2e, sW2o, sb2o,
                                             tW1, tb1, tW2e, tb2e, tW2o, tb2o,
                                             out);
}
